// VMambaBranch_59450937312024
// MI455X (gfx1250) — matrix operand order repaired and checked by reading, not compiled, not run on hardware
//
#include <hip/hip_runtime.h>
#include <hip/hip_bf16.h>

#define Hh 14
#define Ww 14
#define Lseq 196
#define EMBED 768
#define DIM   3072
#define DIN   6144
#define DTR   192
#define DST   16
#define DCONV 4
#define DEPTH 2
#define BATCH 2
#define PATCH 16
#define MROWS (BATCH * Lseq) /* 392 logical rows */
#define MP    400            /* padded to 25 full 16-row tiles */

typedef __attribute__((ext_vector_type(16))) _Float16 v16h;
typedef __attribute__((ext_vector_type(8)))  float    v8f;

__device__ __forceinline__ v16h cvt16(float4 a, float4 b, float4 c, float4 d)
{
    v16h v;
    v[0]  = (_Float16)a.x; v[1]  = (_Float16)a.y; v[2]  = (_Float16)a.z; v[3]  = (_Float16)a.w;
    v[4]  = (_Float16)b.x; v[5]  = (_Float16)b.y; v[6]  = (_Float16)b.z; v[7]  = (_Float16)b.w;
    v[8]  = (_Float16)c.x; v[9]  = (_Float16)c.y; v[10] = (_Float16)c.z; v[11] = (_Float16)c.w;
    v[12] = (_Float16)d.x; v[13] = (_Float16)d.y; v[14] = (_Float16)d.z; v[15] = (_Float16)d.w;
    return v;
}

// ---------------------------------------------------------------------------
// WMMA GEMM:  C[MP x N] = A[MP x K] * W[N x K]^T   (fp32 in, f16 WMMA, f32 acc)
// One wave computes a 16x32 slab of C (two 16x16 WMMA tiles sharing the A
// fragment). All row counts are padded to MP so no bounds checks are needed.
// Fragment layouts used by this kernel (wave32, contiguous k order):
//   A lane l: row l&15, K in {ka..ka+15},              ka=(l>>4)*16
//   B lane l: col l&15, K in {kw..kw+15},              kw=(l>>4)*16
//   D vgpr r: row r+8*(l>>4), col l&15
// ---------------------------------------------------------------------------
__global__ __launch_bounds__(32)
void k_gemm_wmma(const float* __restrict__ A, int lda,
                 const float* __restrict__ W, int K,
                 float* __restrict__ C, int ldc)
{
    const int lane = threadIdx.x;
    const int half = lane >> 4;
    const int lm   = lane & 15;
    const int m    = blockIdx.y * 16 + lm;
    const int n0   = blockIdx.x * 32 + lm;
    const int ka   = half * 16;
    const int kw   = half * 16;

    const float* arow  = A + (size_t)m * lda;
    const float* wrow0 = W + (size_t)n0 * K;
    const float* wrow1 = wrow0 + (size_t)16 * K;

    v8f acc0 = {}, acc1 = {};
    for (int k0 = 0; k0 < K; k0 += 32) {
        __builtin_prefetch(arow  + k0 + ka + 32, 0, 0);
        __builtin_prefetch(wrow0 + k0 + kw + 32, 0, 0);
        __builtin_prefetch(wrow1 + k0 + kw + 32, 0, 0);

        float4 a0 = *(const float4*)(arow + k0 + ka);
        float4 a1 = *(const float4*)(arow + k0 + ka + 4);
        float4 a2 = *(const float4*)(arow + k0 + ka + 8);
        float4 a3 = *(const float4*)(arow + k0 + ka + 12);
        v16h af = cvt16(a0, a1, a2, a3);

        float4 w0 = *(const float4*)(wrow0 + k0 + kw);
        float4 w1 = *(const float4*)(wrow0 + k0 + kw + 4);
        float4 w2 = *(const float4*)(wrow0 + k0 + kw + 8);
        float4 w3 = *(const float4*)(wrow0 + k0 + kw + 12);
        v16h bf0 = cvt16(w0, w1, w2, w3);
        acc0 = __builtin_amdgcn_wmma_f32_16x16x32_f16(
                   false, af, false, bf0, (short)0, acc0, false, false);

        float4 u0 = *(const float4*)(wrow1 + k0 + kw);
        float4 u1 = *(const float4*)(wrow1 + k0 + kw + 4);
        float4 u2 = *(const float4*)(wrow1 + k0 + kw + 8);
        float4 u3 = *(const float4*)(wrow1 + k0 + kw + 12);
        v16h bf1 = cvt16(u0, u1, u2, u3);
        acc1 = __builtin_amdgcn_wmma_f32_16x16x32_f16(
                   false, af, false, bf1, (short)0, acc1, false, false);
    }

#pragma unroll
    for (int r = 0; r < 8; ++r) {
        size_t mm = (size_t)(blockIdx.y * 16 + r + half * 8);
        C[mm * ldc + n0]      = acc0[r];
        C[mm * ldc + n0 + 16] = acc1[r];
    }
}

// ---------------------------------------------------------------------------
// Device-side replication of the Python spiral / radial / boundary orderings.
// ---------------------------------------------------------------------------
__global__ void k_build_perm(int* __restrict__ perm)
{
    if (threadIdx.x != 0 || blockIdx.x != 0) return;
    const int ch = Hh / 2, cw = Ww / 2;
    bool seen[Lseq];
    for (int i = 0; i < Lseq; ++i) seen[i] = false;
    int cnt = 0;
    const double TWO_PI = 6.283185307179586476925287;
    for (int r = 0; r < (Hh > Ww ? Hh : Ww); ++r) {
        int na = (2 * r > 8) ? 2 * r : 8;
        for (int i = 0; i < na; ++i) {
            double ang = TWO_PI * (double)i / (double)na;
            int h = (int)((double)ch + (double)r * cos(ang));
            int w = (int)((double)cw + (double)r * sin(ang));
            if (h >= 0 && h < Hh && w >= 0 && w < Ww) {
                int idx = h * Ww + w;
                if (!seen[idx]) { seen[idx] = true; perm[cnt++] = idx; }
            }
        }
    }
    for (int i = 0; i < Lseq; ++i) if (!seen[i]) perm[cnt++] = i;

    double dd[Lseq], aa[Lseq];
    int ord[Lseq];
    for (int h = 0; h < Hh; ++h)
        for (int w = 0; w < Ww; ++w) {
            int i = h * Ww + w;
            dd[i] = sqrt((double)((h - ch) * (h - ch) + (w - cw) * (w - cw)));
            aa[i] = atan2((double)(h - ch), (double)(w - cw));
            ord[i] = i;
        }
    for (int i = 1; i < Lseq; ++i) {
        int oi = ord[i];
        double di = dd[oi], ai = aa[oi];
        int j = i - 1;
        while (j >= 0) {
            int oj = ord[j];
            bool after = (dd[oj] < di) || (dd[oj] == di && aa[oj] > ai);
            if (!after) break;
            ord[j + 1] = oj; --j;
        }
        ord[j + 1] = oi;
    }
    for (int i = 0; i < Lseq; ++i) perm[Lseq + i] = ord[i];

    int bc = 0;
    for (int h = 0; h < Hh; ++h)
        for (int w = 0; w < Ww; ++w)
            if (h == 0 || h == Hh - 1 || w == 0 || w == Ww - 1)
                perm[2 * Lseq + bc++] = h * Ww + w;
    for (int h = 0; h < Hh; ++h)
        for (int w = 0; w < Ww; ++w)
            if (!(h == 0 || h == Hh - 1 || w == 0 || w == Ww - 1))
                perm[2 * Lseq + bc++] = h * Ww + w;
}

// ---------------------------------------------------------------------------
__global__ void k_patch_embed(const float* __restrict__ x,
                              const float* __restrict__ abcde,
                              const float* __restrict__ pw,
                              const float* __restrict__ pb,
                              const float* __restrict__ pos,
                              const float* __restrict__ aw,
                              const float* __restrict__ ab,
                              float* __restrict__ xe)
{
    int gid = blockIdx.x * blockDim.x + threadIdx.x;
    if (gid >= MROWS * EMBED) return;
    int e = gid % EMBED;
    int l = (gid / EMBED) % Lseq;
    int b = gid / (EMBED * Lseq);
    int lh = l / Ww, lw = l % Ww;
    float s = pb[e];
    for (int c = 0; c < 3; ++c)
        for (int ph = 0; ph < PATCH; ++ph) {
            const float* xrow = x + (((size_t)(b * 3 + c) * 224) + (lh * PATCH + ph)) * 224
                                  + lw * PATCH;
            const float* wrow = pw + (((size_t)(e * 3 + c) * PATCH) + ph) * PATCH;
            for (int q = 0; q < PATCH; ++q) s += xrow[q] * wrow[q];
        }
    s += pos[(size_t)l * EMBED + e];
    float ap = ab[e];
    for (int j = 0; j < 5; ++j) ap += abcde[b * 5 + j] * aw[e * 5 + j];
    xe[gid] = s + ap;
}

__global__ void k_build_xm(const float* __restrict__ xe,
                           const int* __restrict__ perm,
                           float* __restrict__ xm)
{
    int gid = blockIdx.x * blockDim.x + threadIdx.x;
    if (gid >= MROWS * EMBED) return;
    int e = gid % EMBED;
    int l = (gid / EMBED) % Lseq;
    int b = gid / (EMBED * Lseq);
    const float* xb = xe + (size_t)b * Lseq * EMBED;
    float* row = xm + (size_t)(b * Lseq + l) * DIM;
    row[0 * EMBED + e] = xb[(size_t)perm[l] * EMBED + e];
    row[1 * EMBED + e] = xb[(size_t)perm[Lseq + l] * EMBED + e];
    row[2 * EMBED + e] = xb[(size_t)perm[2 * Lseq + l] * EMBED + e];
    row[3 * EMBED + e] = xb[(size_t)l * EMBED + e];
}

// LayerNorm over last dim D; optional pre-bias added before stats. In-place OK.
__global__ __launch_bounds__(256)
void k_layernorm(const float* __restrict__ x, float* __restrict__ y,
                 const float* __restrict__ w, const float* __restrict__ bb,
                 const float* __restrict__ preb, int D)
{
    __shared__ float s1[256], s2[256];
    int row = blockIdx.x, tid = threadIdx.x;
    const float* xr = x + (size_t)row * D;
    float* yr = y + (size_t)row * D;
    float a = 0.f, q = 0.f;
    for (int i = tid; i < D; i += 256) {
        float v = xr[i] + (preb ? preb[i] : 0.f);
        a += v; q += v * v;
    }
    s1[tid] = a; s2[tid] = q;
    __syncthreads();
    for (int s = 128; s > 0; s >>= 1) {
        if (tid < s) { s1[tid] += s1[tid + s]; s2[tid] += s2[tid + s]; }
        __syncthreads();
    }
    float mean = s1[0] / (float)D;
    float var  = s2[0] / (float)D - mean * mean;
    float rstd = rsqrtf(var + 1e-5f);
    for (int i = tid; i < D; i += 256) {
        float v = xr[i] + (preb ? preb[i] : 0.f);
        yr[i] = (v - mean) * rstd * w[i] + bb[i];
    }
}

// Causal depthwise conv (K=4, pad 3 left) + bias + SiLU.
__global__ void k_conv_silu(const float* __restrict__ xr, int ldr,
                            const float* __restrict__ cw,
                            const float* __restrict__ cb,
                            float* __restrict__ xc)
{
    int gid = blockIdx.x * blockDim.x + threadIdx.x;
    if (gid >= MROWS * DIN) return;
    int d = gid % DIN;
    int t = (gid / DIN) % Lseq;
    int b = gid / (DIN * Lseq);
    float s = cb[d];
#pragma unroll
    for (int j = 0; j < DCONV; ++j) {
        int tt = t - (DCONV - 1) + j;
        if (tt >= 0)
            s += xr[(size_t)(b * Lseq + tt) * ldr + d] * cw[d * DCONV + j];
    }
    xc[gid] = s / (1.f + __expf(-s));
}

__global__ void k_softplus_bias(float* __restrict__ dt,
                                const float* __restrict__ bias, int n)
{
    int gid = blockIdx.x * blockDim.x + threadIdx.x;
    if (gid >= n) return;
    float v = dt[gid] + bias[gid % DIN];
    dt[gid] = (v > 20.f) ? v : log1pf(__expf(v));
}

// Selective scan: one thread per (b, d); 16-wide state in registers.
__global__ void k_scan(const float* __restrict__ xc,
                       const float* __restrict__ dbl,
                       const float* __restrict__ dt,
                       const float* __restrict__ res, // xr + DIN, row stride 2*DIN
                       const float* __restrict__ A_log,
                       const float* __restrict__ Dp,
                       float* __restrict__ yb)
{
    int gid = blockIdx.x * blockDim.x + threadIdx.x;
    if (gid >= BATCH * DIN) return;
    int d = gid % DIN;
    int b = gid / DIN;
    float Ar[DST], st[DST];
#pragma unroll
    for (int n = 0; n < DST; ++n) {
        Ar[n] = -__expf(A_log[(size_t)d * DST + n]);
        st[n] = 0.f;
    }
    float Dd = Dp[d];
    for (int t = 0; t < Lseq; ++t) {
        size_t row = (size_t)(b * Lseq + t);
        float dtv = dt[row * DIN + d];
        float xv  = xc[row * DIN + d];
        const float* bc = dbl + row * (DTR + 2 * DST) + DTR;
        float y = 0.f;
#pragma unroll
        for (int n = 0; n < DST; ++n) {
            st[n] = st[n] * __expf(dtv * Ar[n]) + dtv * bc[n] * xv;
            y += st[n] * bc[DST + n];
        }
        float rv  = res[row * (2 * DIN) + d];
        float sil = rv / (1.f + __expf(-rv));
        yb[row * DIN + d] = (y + xv * Dd) * sil;
    }
}

__global__ void k_add(float* __restrict__ a, const float* __restrict__ b, int n)
{
    int gid = blockIdx.x * blockDim.x + threadIdx.x;
    if (gid < n) a[gid] += b[gid];
}

__global__ void k_zero(float* __restrict__ p, int n)
{
    int gid = blockIdx.x * blockDim.x + threadIdx.x;
    if (gid < n) p[gid] = 0.f;
}

__global__ void k_mean(const float* __restrict__ xpj, float* __restrict__ out)
{
    int gid = blockIdx.x * blockDim.x + threadIdx.x;
    if (gid >= BATCH * EMBED) return;
    int e = gid % EMBED;
    int b = gid / EMBED;
    float s = 0.f;
    for (int l = 0; l < Lseq; ++l)
        s += xpj[(size_t)(b * Lseq + l) * EMBED + e];
    out[gid] = s / (float)Lseq;
}

// ---------------------------------------------------------------------------
extern "C" void kernel_launch(void* const* d_in, const int* in_sizes, int n_in,
                              void* d_out, int out_size, void* d_ws, size_t ws_size,
                              hipStream_t stream)
{
    (void)in_sizes; (void)n_in; (void)out_size; (void)ws_size;
    const float* x         = (const float*)d_in[0];
    const float* abcde     = (const float*)d_in[1];
    const float* patch_w   = (const float*)d_in[2];
    const float* patch_b   = (const float*)d_in[3];
    const float* pos       = (const float*)d_in[4];
    const float* abcde_w   = (const float*)d_in[5];
    const float* abcde_b   = (const float*)d_in[6];
    const float* ln_w      = (const float*)d_in[7];
    const float* ln_b      = (const float*)d_in[8];
    const float* in_proj_w = (const float*)d_in[9];
    const float* conv_w    = (const float*)d_in[10];
    const float* conv_b    = (const float*)d_in[11];
    const float* x_proj_w  = (const float*)d_in[12];
    const float* dt_proj_w = (const float*)d_in[13];
    const float* A_log     = (const float*)d_in[14];
    const float* D_param   = (const float*)d_in[15];
    const float* out_proj_w= (const float*)d_in[16];
    const float* proj_w    = (const float*)d_in[17];
    const float* proj_b    = (const float*)d_in[18];
    const float* norm_w    = (const float*)d_in[19];
    const float* norm_b    = (const float*)d_in[20];
    const float* dt_proj_b = (const float*)d_in[21];
    float* out = (float*)d_out;

    char* ws = (char*)d_ws;
    size_t off = 0;
    auto alloc = [&](size_t bytes) {
        size_t o = off;
        off += (bytes + 255) & ~(size_t)255;
        return o;
    };
    // All GEMM-visible matrices padded to MP rows (pad rows kept at zero).
    int*   perm = (int*)  (ws + alloc((size_t)3 * Lseq * sizeof(int)));
    float* xe   = (float*)(ws + alloc((size_t)MP * EMBED * 4));
    float* xm   = (float*)(ws + alloc((size_t)MP * DIM * 4));
    float* xn   = (float*)(ws + alloc((size_t)MP * DIM * 4));   // LN out; reused for out_proj result
    float* xr   = (float*)(ws + alloc((size_t)MP * 2 * DIN * 4));
    float* xc   = (float*)(ws + alloc((size_t)MP * DIN * 4));
    float* dbl  = (float*)(ws + alloc((size_t)MP * (DTR + 2 * DST) * 4));
    float* dtb  = (float*)(ws + alloc((size_t)MP * DIN * 4));
    float* yb   = (float*)(ws + alloc((size_t)MP * DIN * 4));
    float* xpj  = xe; // xe no longer needed after k_build_xm

    const int MT  = MP / 16;      // 25 M-tiles
    const int PAD = MP - MROWS;   // 8 pad rows

    // Zero pad rows of every buffer used as a GEMM A-operand.
    k_zero<<<(PAD * DIM + 255) / 256, 256, 0, stream>>>(xn + (size_t)MROWS * DIM, PAD * DIM);
    k_zero<<<(PAD * DIM + 255) / 256, 256, 0, stream>>>(xm + (size_t)MROWS * DIM, PAD * DIM);
    k_zero<<<(PAD * DIN + 255) / 256, 256, 0, stream>>>(xc + (size_t)MROWS * DIN, PAD * DIN);
    k_zero<<<(PAD * DIN + 255) / 256, 256, 0, stream>>>(yb + (size_t)MROWS * DIN, PAD * DIN);
    k_zero<<<(PAD * (DTR + 2 * DST) + 255) / 256, 256, 0, stream>>>(
        dbl + (size_t)MROWS * (DTR + 2 * DST), PAD * (DTR + 2 * DST));

    k_build_perm<<<1, 1, 0, stream>>>(perm);
    {
        int tot = MROWS * EMBED;
        k_patch_embed<<<(tot + 255) / 256, 256, 0, stream>>>(
            x, abcde, patch_w, patch_b, pos, abcde_w, abcde_b, xe);
        k_build_xm<<<(tot + 255) / 256, 256, 0, stream>>>(xe, perm, xm);
    }

    for (int l = 0; l < DEPTH; ++l) {
        k_layernorm<<<MROWS, 256, 0, stream>>>(
            xm, xn, ln_w + l * DIM, ln_b + l * DIM, nullptr, DIM);

        // xr = xn @ in_proj_w[l].T     (MP x 12288, K=3072)
        k_gemm_wmma<<<dim3(2 * DIN / 32, MT), 32, 0, stream>>>(
            xn, DIM, in_proj_w + (size_t)l * 2 * DIN * DIM, DIM, xr, 2 * DIN);

        {
            int tot = MROWS * DIN;
            k_conv_silu<<<(tot + 255) / 256, 256, 0, stream>>>(
                xr, 2 * DIN, conv_w + (size_t)l * DIN * DCONV, conv_b + l * DIN, xc);
        }

        // dbl = xc @ x_proj_w[l].T     (MP x 224, K=6144)
        k_gemm_wmma<<<dim3((DTR + 2 * DST) / 32, MT), 32, 0, stream>>>(
            xc, DIN, x_proj_w + (size_t)l * (DTR + 2 * DST) * DIN, DIN,
            dbl, DTR + 2 * DST);

        // dtb = dbl[:, :192] @ dt_proj_w[l].T   (MP x 6144, K=192)
        k_gemm_wmma<<<dim3(DIN / 32, MT), 32, 0, stream>>>(
            dbl, DTR + 2 * DST, dt_proj_w + (size_t)l * DIN * DTR, DTR,
            dtb, DIN);

        {
            int tot = MROWS * DIN;
            k_softplus_bias<<<(tot + 255) / 256, 256, 0, stream>>>(
                dtb, dt_proj_b + l * DIN, tot);
        }
        {
            int tot = BATCH * DIN;
            k_scan<<<(tot + 255) / 256, 256, 0, stream>>>(
                xc, dbl, dtb, xr + DIN,
                A_log + (size_t)l * DIN * DST, D_param + l * DIN, yb);
        }

        // xn = yb @ out_proj_w[l].T    (MP x 3072, K=6144)
        k_gemm_wmma<<<dim3(DIM / 32, MT), 32, 0, stream>>>(
            yb, DIN, out_proj_w + (size_t)l * DIM * DIN, DIN, xn, DIM);

        {
            int tot = MROWS * DIM;
            k_add<<<(tot + 255) / 256, 256, 0, stream>>>(xm, xn, tot);
        }
    }

    // xpj = xm @ proj_w.T  (MP x 768, K=3072), then +proj_b, LN, mean over L
    k_gemm_wmma<<<dim3(EMBED / 32, MT), 32, 0, stream>>>(
        xm, DIM, proj_w, DIM, xpj, EMBED);
    k_layernorm<<<MROWS, 256, 0, stream>>>(xpj, xpj, norm_w, norm_b, proj_b, EMBED);
    {
        int tot = BATCH * EMBED;
        k_mean<<<(tot + 255) / 256, 256, 0, stream>>>(xpj, out);
    }
}
